// QStack_15161234554907
// MI455X (gfx1250) — matrix operand order repaired and checked by reading, not compiled, not run on hardware
//
#include <hip/hip_runtime.h>
#include <cmath>

typedef __attribute__((ext_vector_type(16))) __bf16 v16bf;
typedef __attribute__((ext_vector_type(8)))  float  v8f;

#define K_CB   2
#define DIM    128
#define NEMB   4096
#define BATCH  32
#define HW     1024
#define NROWS  (BATCH * HW)      // 32768 rows per codebook
#define CHUNK  32                // codebook columns per LDS buffer
#define NCHUNK (NEMB / CHUNK)    // 128 chunks
#define BPITCH 136               // LDS pitch (bf16): 128 + 8 pad -> conflict-free ds reads

union Frag { v16bf v; uint4 u[2]; };

// ---------------------------------------------------------------------------
// Zero accumulators each call (harness does not re-poison between replays).
// ---------------------------------------------------------------------------
__global__ void init_kernel(float* __restrict__ hist, float* __restrict__ diffsum) {
    int t = blockIdx.x * blockDim.x + threadIdx.x;
    if (t < K_CB * NEMB) hist[t] = 0.0f;
    if (t < K_CB)        diffsum[t] = 0.0f;
}

// ---------------------------------------------------------------------------
// x [B, 256, 32, 32] fp32  ->  A_hi/A_lo [K][N=32768][128] bf16 (row-major)
// ---------------------------------------------------------------------------
__global__ __launch_bounds__(256) void prep_x_kernel(const float* __restrict__ x,
                                                     __bf16* __restrict__ Ahi,
                                                     __bf16* __restrict__ Alo) {
    __shared__ float tile[32][33];
    int bz = blockIdx.z;                  // b * K + k
    int b = bz >> 1, k = bz & 1;
    int hw0 = blockIdx.x * 32;
    int d0  = blockIdx.y * 32;
    int tx = threadIdx.x, ty = threadIdx.y;   // 32 x 8

    #pragma unroll
    for (int j = 0; j < 4; ++j) {
        int d = d0 + ty + j * 8;
        tile[ty + j * 8][tx] =
            x[((size_t)b * 256 + k * 128 + d) * (size_t)HW + hw0 + tx];
    }
    __syncthreads();
    #pragma unroll
    for (int j = 0; j < 4; ++j) {
        int hw = hw0 + ty + j * 8;
        size_t n = (size_t)k * NROWS + (size_t)b * HW + hw;
        float v = tile[tx][ty + j * 8];
        __bf16 h = (__bf16)v;
        __bf16 l = (__bf16)(v - (float)h);
        Ahi[n * DIM + d0 + tx] = h;
        Alo[n * DIM + d0 + tx] = l;
    }
}

// ---------------------------------------------------------------------------
// embed [K][128][4096] -> B_hi/B_lo [K][4096][128] bf16, ET fp32, 0.5*||e||^2
// ---------------------------------------------------------------------------
__global__ __launch_bounds__(256) void prep_e_kernel(const float* __restrict__ embed,
                                                     __bf16* __restrict__ Bhi,
                                                     __bf16* __restrict__ Blo,
                                                     float*  __restrict__ ET,
                                                     float*  __restrict__ halfnorm) {
    int k = blockIdx.y;
    int j = blockIdx.x * blockDim.x + threadIdx.x;    // 0..4095
    const float* E = embed + (size_t)k * DIM * NEMB;
    size_t rowoff = ((size_t)k * NEMB + j) * DIM;
    float nrm = 0.0f;
    for (int d = 0; d < DIM; ++d) {
        float e = E[(size_t)d * NEMB + j];
        nrm += e * e;
        ET[rowoff + d] = e;
        __bf16 h = (__bf16)e;
        Bhi[rowoff + d] = h;
        Blo[rowoff + d] = (__bf16)(e - (float)h);
    }
    halfnorm[k * NEMB + j] = 0.5f * nrm;
}

// ---------------------------------------------------------------------------
// Fused split-bf16 GEMM + argmin with ASYNC double-buffered B staging.
// Block: 8 waves x 16 rows. B chunks (32 cols, hi+lo) streamed into LDS via
// global_load_async_to_lds_b128 (ASYNCcnt), one chunk in flight ahead of
// compute. All 4096 halfnorms parked in LDS up front.
// ---------------------------------------------------------------------------
__global__ __launch_bounds__(256) void gemm_argmin_kernel(
        const __bf16* __restrict__ Ahi, const __bf16* __restrict__ Alo,
        const __bf16* __restrict__ Bhi, const __bf16* __restrict__ Blo,
        const float*  __restrict__ halfnorm,
        int* __restrict__ idxbuf, float* __restrict__ out_arg) {
    __shared__ __align__(16) __bf16 sB[2][2][CHUNK * BPITCH]; // [buf][hi/lo]
    __shared__ __align__(16) float  sHnAll[NEMB];

    int k    = blockIdx.y;
    int t    = threadIdx.x;
    int wave = t >> 5, lane = t & 31;
    int n0   = blockIdx.x * 128 + wave * 16;
    int rowl = lane & 15;
    int grp  = lane >> 4;

    const __bf16* gBhi = Bhi + (size_t)k * NEMB * DIM;
    const __bf16* gBlo = Blo + (size_t)k * NEMB * DIM;

    // ---- park all halfnorms for this codebook in LDS (16 KB, one time) ----
    {
        const uint4* src = (const uint4*)(halfnorm + k * NEMB);
        uint4* dst = (uint4*)sHnAll;
        #pragma unroll
        for (int i = 0; i < 4; ++i) dst[t + i * 256] = src[t + i * 256];
    }

    // ---- A fragments resident in VGPRs (ISA 16-bit 16x32 A layout) ----
    Frag ahi[4], alo[4];
    {
        const __bf16* arh = Ahi + ((size_t)k * NROWS + n0 + rowl) * DIM;
        const __bf16* arl = Alo + ((size_t)k * NROWS + n0 + rowl) * DIM;
        int ko = grp * 8;
        #pragma unroll
        for (int kc4 = 0; kc4 < 4; ++kc4) {
            int kc = kc4 * 32;
            ahi[kc4].u[0] = *(const uint4*)(arh + kc + ko);
            ahi[kc4].u[1] = *(const uint4*)(arh + kc + 16 + ko);
            alo[kc4].u[0] = *(const uint4*)(arl + kc + ko);
            alo[kc4].u[1] = *(const uint4*)(arl + kc + 16 + ko);
        }
    }

    // ---- async staging setup: 4 x b128 per wave per chunk ----
    // thread -> (col cl, 16B sub-chunk c16); global row pitch 256 B
    unsigned c16   = (t & 15) * 16u;
    unsigned cl0   = (unsigned)(t >> 4);          // 0..15
    unsigned goff0 = cl0 * 256u + c16;
    unsigned goff1 = goff0 + 16u * 256u;          // cols 16..31
    unsigned lrel0 = cl0 * (BPITCH * 2u) + c16;   // LDS bytes within a plane
    unsigned lrel1 = lrel0 + 16u * (BPITCH * 2u);
    // low 32 bits of a flat pointer into LDS == LDS byte offset
    unsigned hiB[2], loB[2];
    hiB[0] = (unsigned)(uintptr_t)&sB[0][0][0];
    loB[0] = (unsigned)(uintptr_t)&sB[0][1][0];
    hiB[1] = (unsigned)(uintptr_t)&sB[1][0][0];
    loB[1] = (unsigned)(uintptr_t)&sB[1][1][0];

    auto stage = [&](int buf, int chunk) {
        const char* srcH = (const char*)gBhi + (size_t)chunk * (CHUNK * 256);
        const char* srcL = (const char*)gBlo + (size_t)chunk * (CHUNK * 256);
        asm volatile("global_load_async_to_lds_b128 %0, %1, %2 offset:0"
                     :: "v"(hiB[buf] + lrel0), "v"(goff0), "s"(srcH) : "memory");
        asm volatile("global_load_async_to_lds_b128 %0, %1, %2 offset:0"
                     :: "v"(hiB[buf] + lrel1), "v"(goff1), "s"(srcH) : "memory");
        asm volatile("global_load_async_to_lds_b128 %0, %1, %2 offset:0"
                     :: "v"(loB[buf] + lrel0), "v"(goff0), "s"(srcL) : "memory");
        asm volatile("global_load_async_to_lds_b128 %0, %1, %2 offset:0"
                     :: "v"(loB[buf] + lrel1), "v"(goff1), "s"(srcL) : "memory");
    };

    float bestv[8];
    int   besti[8];
    #pragma unroll
    for (int r = 0; r < 8; ++r) { bestv[r] = -3.402823e38f; besti[r] = 0; }

    int koB = grp * 8;

    stage(0, 0);
    int cur = 0;
    for (int c = 0; c < NCHUNK; ++c) {
        if (c + 1 < NCHUNK) {
            stage(cur ^ 1, c + 1);
            asm volatile("s_wait_asynccnt 4" ::: "memory");   // chunk c landed (this wave)
        } else {
            asm volatile("s_wait_asynccnt 0" ::: "memory");
        }
        __syncthreads();                                      // chunk c landed (all waves)

        #pragma unroll
        for (int ct = 0; ct < 2; ++ct) {
            v8f acc = {0.f, 0.f, 0.f, 0.f, 0.f, 0.f, 0.f, 0.f};
            const __bf16* bh = &sB[cur][0][(ct * 16 + rowl) * BPITCH];
            const __bf16* bl = &sB[cur][1][(ct * 16 + rowl) * BPITCH];
            #pragma unroll
            for (int kc4 = 0; kc4 < 4; ++kc4) {
                int kc = kc4 * 32;
                Frag fbh, fbl;
                fbh.u[0] = *(const uint4*)(bh + kc + koB);
                fbh.u[1] = *(const uint4*)(bh + kc + koB + 16);
                fbl.u[0] = *(const uint4*)(bl + kc + koB);
                fbl.u[1] = *(const uint4*)(bl + kc + koB + 16);
                // S ~= Ahi*Bhi + Alo*Bhi + Ahi*Blo (near-fp32)
                acc = __builtin_amdgcn_wmma_f32_16x16x32_bf16(
                        false, ahi[kc4].v, false, fbh.v, (short)0, acc, false, false);
                acc = __builtin_amdgcn_wmma_f32_16x16x32_bf16(
                        false, alo[kc4].v, false, fbh.v, (short)0, acc, false, false);
                acc = __builtin_amdgcn_wmma_f32_16x16x32_bf16(
                        false, ahi[kc4].v, false, fbl.v, (short)0, acc, false, false);
            }
            int   col = c * CHUNK + ct * 16 + rowl;
            float hn  = sHnAll[col];
            #pragma unroll
            for (int r = 0; r < 8; ++r) {
                float s = acc[r] - hn;                // argmax(f.e - 0.5||e||^2)
                if (s > bestv[r]) { bestv[r] = s; besti[r] = col; }
            }
        }
        __syncthreads();      // all waves done reading buf 'cur' before reuse
        cur ^= 1;
    }

    // reduce across the 16 lanes sharing each row (xor<16 stays in half-group)
    #pragma unroll
    for (int off = 1; off < 16; off <<= 1) {
        #pragma unroll
        for (int r = 0; r < 8; ++r) {
            float ov = __shfl_xor(bestv[r], off, 32);
            int   oi = __shfl_xor(besti[r], off, 32);
            if (ov > bestv[r] || (ov == bestv[r] && oi < besti[r])) {
                bestv[r] = ov; besti[r] = oi;
            }
        }
    }
    if (rowl == 0) {                      // lanes 0 / 16 hold rows r / r+8
        #pragma unroll
        for (int r = 0; r < 8; ++r) {
            int row = n0 + r + grp * 8;
            idxbuf[k * NROWS + row]  = besti[r];
            out_arg[k * NROWS + row] = (float)besti[r];
        }
    }
}

// ---------------------------------------------------------------------------
// Gather z_q = E[:, idx] (fp32 exact), fused MSE + histogram.
// ---------------------------------------------------------------------------
__global__ __launch_bounds__(256) void gather_kernel(
        const float* __restrict__ x, const float* __restrict__ ET,
        const int* __restrict__ idxbuf,
        float* __restrict__ zq, float* __restrict__ hist,
        float* __restrict__ diffsum) {
    __shared__ float qtile[64 * 132];
    __shared__ int   sidx[64];
    __shared__ float red[256];
    int k  = blockIdx.y;
    int t  = threadIdx.x;
    int n0 = blockIdx.x * 64;

    if (t < 64) {
        int id = idxbuf[k * NROWS + n0 + t];
        sidx[t] = id;
        atomicAdd(&hist[k * NEMB + id], 1.0f);
    }
    __syncthreads();
    #pragma unroll
    for (int i = 0; i < 8; ++i) {
        int v  = t + i * 256;
        int rl = v >> 5, c4 = v & 31;
        uint4 q = *((const uint4*)(ET + ((size_t)k * NEMB + sidx[rl]) * DIM) + c4);
        *((uint4*)(&qtile[rl * 132]) + c4) = q;
    }
    __syncthreads();

    int b = n0 >> 10, hw0 = n0 & 1023;
    int nl = t & 63, dq = t >> 6;
    float acc = 0.0f;
    #pragma unroll 4
    for (int i = 0; i < 32; ++i) {
        int d = dq + i * 4;
        size_t go = ((size_t)b * 256 + k * 128 + d) * (size_t)HW + hw0 + nl;
        float q = qtile[nl * 132 + d];
        float z = x[go];
        zq[go] = q;
        float df = q - z;
        acc += df * df;
    }
    red[t] = acc;
    __syncthreads();
    for (int s = 128; s > 0; s >>= 1) {
        if (t < s) red[t] += red[t + s];
        __syncthreads();
    }
    if (t == 0) atomicAdd(&diffsum[k], red[0]);
}

// ---------------------------------------------------------------------------
__global__ __launch_bounds__(256) void finalize_kernel(
        const float* __restrict__ hist, const float* __restrict__ diffsum,
        float* __restrict__ out_diff, float* __restrict__ out_ppl) {
    __shared__ float red[256];
    int k = blockIdx.x, t = threadIdx.x;
    float s = 0.0f;
    for (int i = t; i < NEMB; i += 256) {
        float p = hist[k * NEMB + i] * (1.0f / (float)NROWS);
        s += p * logf(p + 1e-10f);
    }
    red[t] = s;
    __syncthreads();
    for (int st = 128; st > 0; st >>= 1) {
        if (t < st) red[t] += red[t + st];
        __syncthreads();
    }
    if (t == 0) {
        out_ppl[k]  = expf(-red[0]);
        out_diff[k] = diffsum[k] * (1.0f / ((float)NROWS * (float)DIM));
    }
}

// ---------------------------------------------------------------------------
extern "C" void kernel_launch(void* const* d_in, const int* in_sizes, int n_in,
                              void* d_out, int out_size, void* d_ws, size_t ws_size,
                              hipStream_t stream) {
    const float* x     = (const float*)d_in[0];   // [32,256,32,32]
    const float* embed = (const float*)d_in[1];   // [2,128,4096]

    float* out      = (float*)d_out;
    float* out_zq   = out;                        // 8,388,608
    float* out_diff = out + 8388608;              // 2
    float* out_arg  = out + 8388610;              // 65,536 (float-cast indices)
    float* out_ppl  = out + 8388610 + 65536;      // 2

    char*  ws  = (char*)d_ws;
    size_t off = 0;
    auto carve = [&](size_t bytes) {
        void* p = ws + off;
        off += (bytes + 255) & ~(size_t)255;
        return p;
    };
    __bf16* Ahi      = (__bf16*)carve((size_t)K_CB * NROWS * DIM * 2);
    __bf16* Alo      = (__bf16*)carve((size_t)K_CB * NROWS * DIM * 2);
    __bf16* Bhi      = (__bf16*)carve((size_t)K_CB * NEMB  * DIM * 2);
    __bf16* Blo      = (__bf16*)carve((size_t)K_CB * NEMB  * DIM * 2);
    float*  ET       = (float*) carve((size_t)K_CB * NEMB  * DIM * 4);
    float*  halfnorm = (float*) carve((size_t)K_CB * NEMB * 4);
    int*    idxbuf   = (int*)   carve((size_t)K_CB * NROWS * 4);
    float*  hist     = (float*) carve((size_t)K_CB * NEMB * 4);
    float*  diffsum  = (float*) carve((size_t)K_CB * 4);

    init_kernel<<<32, 256, 0, stream>>>(hist, diffsum);
    prep_x_kernel<<<dim3(32, 4, 64), dim3(32, 8), 0, stream>>>(x, Ahi, Alo);
    prep_e_kernel<<<dim3(16, 2), 256, 0, stream>>>(embed, Bhi, Blo, ET, halfnorm);
    gemm_argmin_kernel<<<dim3(256, 2), 256, 0, stream>>>(Ahi, Alo, Bhi, Blo,
                                                         halfnorm, idxbuf, out_arg);
    gather_kernel<<<dim3(512, 2), 256, 0, stream>>>(x, ET, idxbuf, out_zq,
                                                    hist, diffsum);
    finalize_kernel<<<2, 256, 0, stream>>>(hist, diffsum, out_diff, out_ppl);
}
